// MultiHeadAttention_91010357002474
// MI455X (gfx1250) — matrix operand order repaired and checked by reading, not compiled, not run on hardware
//
#include <hip/hip_runtime.h>
#include <hip/hip_bf16.h>

// CDNA5 / gfx1250 wave32 WMMA implementation of multi-head attention.
// Heavy math: v_wmma_f32_16x16x32_bf16 (fp32 accumulate).
// Attention K/V^T staged through LDS with async global->LDS copies
// (ASYNCcnt-tracked, double buffered) shared by all 8 waves of a block.

typedef __attribute__((ext_vector_type(16))) __bf16 v16bf;
typedef __attribute__((ext_vector_type(8)))  float  v8f;
typedef int i32x4 __attribute__((vector_size(16)));

#define WMMA_BF16(a,b,c) __builtin_amdgcn_wmma_f32_16x16x32_bf16(false,(a),false,(b),(short)0,(c),false,false)

static constexpr int Nb = 2, Lq = 2048, Ss = 2048, Dd = 128, Hh = 32;
static constexpr float SCALE = 0.0883883476483184f;   // 1/sqrt(128)
static constexpr float LOG2E = 1.4426950408889634f;

// ---- CDNA5 async global->LDS staging (guarded; fallback = load + ds_store) --
#if defined(__has_builtin)
#if __has_builtin(__builtin_amdgcn_global_load_async_to_lds_b128) && \
    __has_builtin(__builtin_amdgcn_s_wait_asynccnt)
#define HAVE_ASYNC_LDS 1
#endif
#endif

#ifdef HAVE_ASYNC_LDS
__device__ __forceinline__ void copy16_g2l(const __bf16* g, __bf16* l) {
  __builtin_amdgcn_global_load_async_to_lds_b128(
      (__attribute__((address_space(1))) i32x4*)g,
      (__attribute__((address_space(3))) i32x4*)l, 0, 0);
}
#define WAIT_ASYNC(n) __builtin_amdgcn_s_wait_asynccnt(n)
#else
__device__ __forceinline__ void copy16_g2l(const __bf16* g, __bf16* l) {
  *(uint4*)l = *(const uint4*)g;
}
#define WAIT_ASYNC(n)
#endif

union Frag { v16bf v; uint4 q[2]; unsigned u[8]; };

__device__ __forceinline__ v16bf ld_frag2(const __bf16* p0, const __bf16* p1) {
  Frag f;
  f.q[0] = *(const uint4*)p0;
  f.q[1] = *(const uint4*)p1;
  return f.v;
}

// swap with partner lane (lane ^ 16) inside the wave32
__device__ __forceinline__ float swz_x16(float x) {
  return __builtin_bit_cast(float, __builtin_amdgcn_ds_swizzle(__builtin_bit_cast(int, x), 0x401f));
}
__device__ __forceinline__ unsigned swz_x16u(unsigned x) {
  return (unsigned)__builtin_amdgcn_ds_swizzle((int)x, 0x401f);
}
__device__ __forceinline__ unsigned pk_bf16(float a, float b) {
  unsigned short ua = __builtin_bit_cast(unsigned short, (__bf16)a);
  unsigned short ub = __builtin_bit_cast(unsigned short, (__bf16)b);
  return (unsigned)ua | ((unsigned)ub << 16);
}

// ---------------------------------------------------------------------------
// Kernel 0: fp32 -> bf16 conversions + weight transposes.
// All five arrays have exactly 524288 elements.
//   WkT[h][k][d] = Wk[h][d][k], WvT likewise, Wct[c][r] = Wc[r][c]
// ---------------------------------------------------------------------------
__global__ void cvt_kernel(const float* __restrict__ q,  const float* __restrict__ s,
                           const float* __restrict__ wk, const float* __restrict__ wv,
                           const float* __restrict__ wc,
                           __bf16* __restrict__ qb,  __bf16* __restrict__ sb,
                           __bf16* __restrict__ wkt, __bf16* __restrict__ wvt,
                           __bf16* __restrict__ wct) {
  int i = blockIdx.x * blockDim.x + threadIdx.x;   // 0 .. 524287
  qb[i] = (__bf16)q[i];
  sb[i] = (__bf16)s[i];
  {
    int h = i >> 14, r = i & 16383, d = r >> 7, k = r & 127;
    wkt[(h << 14) + (k << 7) + d] = (__bf16)wk[i];
    wvt[(h << 14) + (k << 7) + d] = (__bf16)wv[i];
  }
  {
    int r = i >> 7, c = i & 127;
    wct[(size_t)c * 4096 + r] = (__bf16)wc[i];
  }
}

// ---------------------------------------------------------------------------
// Kernel 1: per-head projection  P[n,h] = states[n] @ W[h] + b[h]
// One wave computes a 16(s) x 64(k) strip; K=128 contraction in 4 WMMA steps.
// transpose==0: store row-major [S][128] (K matrix)
// transpose==1: store transposed  [128][S] (V^T, contiguous packed store)
// ---------------------------------------------------------------------------
__global__ void proj_kernel(const __bf16* __restrict__ Sb, const __bf16* __restrict__ WT,
                            const float* __restrict__ bias, __bf16* __restrict__ Out,
                            int transpose) {
  const int lane = threadIdx.x & 31, ln = lane & 15, hl = lane >> 4;
  const int wid = blockIdx.x * (blockDim.x >> 5) + (threadIdx.x >> 5);
  const int n = wid >> 13, rem = wid & 8191;
  const int h = rem >> 8, r2 = rem & 255, st = r2 >> 1, k0 = (r2 & 1) * 64;

  const __bf16* srow  = Sb + ((size_t)n * Ss + st * 16 + ln) * Dd;
  const __bf16* wbase = WT + (size_t)h * Dd * Dd;

  v8f acc[4] = {};
  #pragma unroll
  for (int dc = 0; dc < 4; ++dc) {
    v16bf a = ld_frag2(srow + dc * 32 + 16 * hl, srow + dc * 32 + 16 * hl + 8);
    #pragma unroll
    for (int tt = 0; tt < 4; ++tt) {
      const __bf16* bp = wbase + (size_t)(k0 + tt * 16 + ln) * Dd + dc * 32 + 16 * hl;
      v16bf b = ld_frag2(bp, bp + 8);
      acc[tt] = WMMA_BF16(a, b, acc[tt]);
    }
  }

  const size_t nh = (size_t)(n * Hh + h);
  #pragma unroll
  for (int tt = 0; tt < 4; ++tt) {
    const int k = k0 + tt * 16 + ln;
    const float bv = bias[h * Dd + k];
    if (transpose) {
      __bf16* orow = Out + nh * Dd * Ss + (size_t)k * Ss + st * 16 + 8 * hl;
      uint4 w;
      w.x = pk_bf16(acc[tt][0] + bv, acc[tt][1] + bv);
      w.y = pk_bf16(acc[tt][2] + bv, acc[tt][3] + bv);
      w.z = pk_bf16(acc[tt][4] + bv, acc[tt][5] + bv);
      w.w = pk_bf16(acc[tt][6] + bv, acc[tt][7] + bv);
      *(uint4*)orow = w;
    } else {
      __bf16* o = Out + nh * Ss * Dd + (size_t)(st * 16 + 8 * hl) * Dd + k;
      #pragma unroll
      for (int v = 0; v < 8; ++v) o[(size_t)v * Dd] = (__bf16)(acc[tt][v] + bv);
    }
  }
}

// ---------------------------------------------------------------------------
// Kernel 2: flash attention with LDS-staged K / V^T.
// Block = 8 waves = one (n,h) head x 128 queries; each wave owns 16 queries.
// Per 32-wide S-chunk the block async-copies K chunk (8KB, contiguous) and
// V^T chunk (128 rows x 64B) into double-buffered padded LDS; all waves read
// their WMMA fragments from LDS. Online softmax in fp32 with one ds_swizzle
// (xor 16) per lane-pair reduction; P^T B-fragment built with 8 swizzles.
// ---------------------------------------------------------------------------
static constexpr int KPAD = 136;  // 32 rows of 128 bf16, padded (+16B) vs banks
static constexpr int VPAD = 40;   // 128 rows of 32 bf16, padded (+16B)

__global__ void attn_kernel(const __bf16* __restrict__ Qb, const __bf16* __restrict__ Kb,
                            const __bf16* __restrict__ Vt, __bf16* __restrict__ Ctx) {
  __shared__ __bf16 KsBuf[2][32][KPAD];
  __shared__ __bf16 VsBuf[2][128][VPAD];

  const int tid = threadIdx.x;
  const int lane = tid & 31, ln = lane & 15, hl = lane >> 4;
  const int wv = tid >> 5;                  // wave 0..7
  const int b = blockIdx.x;                 // 1024 blocks
  const int n = b >> 9, rem = b & 511, h = rem >> 4, l0 = (rem & 15) * 128 + wv * 16;

  const __bf16* Kbb = Kb + (size_t)(n * Hh + h) * Ss * Dd;
  const __bf16* Vtb = Vt + (size_t)(n * Hh + h) * Dd * Ss;

  // Q B-fragments (B[d][lq] = Q[lq][d]): d = 8*hl+0..7 and 16+8*hl+0..7 per lane
  const __bf16* qrow = Qb + ((size_t)n * Lq + l0 + ln) * Dd;
  v16bf qf[4];
  #pragma unroll
  for (int dc = 0; dc < 4; ++dc) {
    const __bf16* p = qrow + dc * 32 + 8 * hl;
    qf[dc] = ld_frag2(p, p + 16);
  }

  // stage one 32-wide s-chunk into LDS buffer `buf` (4 b128 copies per thread)
  auto stage = [&](int s0, int buf) {
    const __bf16* kg = Kbb + (size_t)s0 * Dd;      // contiguous 8KB
    #pragma unroll
    for (int r = 0; r < 2; ++r) {
      int idx = tid + r * 256;                     // 0..511 b128 pieces
      int row = idx >> 3, part = idx & 7;          // row 0..63, 8x16B per row
      copy16_g2l(kg + idx * 8, &KsBuf[buf][row][part * 8]);
    }
    #pragma unroll
    for (int r = 0; r < 2; ++r) {
      int idx = tid + r * 256;                     // 0..511
      int row = idx >> 2, part = idx & 3;          // row 0..127, 4x16B per row
      copy16_g2l(Vtb + (size_t)row * Ss + s0 + part * 8, &VsBuf[buf][row][part * 8]);
    }
  };

  v8f acc[8] = {};
  float m = -3.0e38f, lsum = 0.f;

  const int NC = Ss / 32;                          // 64 chunks
  stage(0, 0);
  for (int ci = 0; ci < NC; ++ci) {
    const int buf = ci & 1;
    if (ci + 1 < NC) {
      stage((ci + 1) * 32, buf ^ 1);
      WAIT_ASYNC(4);                               // own current-chunk copies done
    } else {
      WAIT_ASYNC(0);
    }
    __syncthreads();                               // whole block's copies visible

    // two 16x16 logit^T tiles (rows=s, cols=lq), contraction over d=128
    v8f lc0 = {}, lc1 = {};
    #pragma unroll
    for (int dc = 0; dc < 4; ++dc) {
      const __bf16* kr0 = &KsBuf[buf][ln][dc * 32 + 8 * hl];
      const __bf16* kr1 = &KsBuf[buf][16 + ln][dc * 32 + 8 * hl];
      lc0 = WMMA_BF16(ld_frag2(kr0, kr0 + 16), qf[dc], lc0);
      lc1 = WMMA_BF16(ld_frag2(kr1, kr1 + 16), qf[dc], lc1);
    }

    // online softmax per query column (lane pair l <-> l^16)
    float x0[8], x1[8], cm = -3.0e38f;
    #pragma unroll
    for (int v = 0; v < 8; ++v) {
      x0[v] = lc0[v] * SCALE;
      x1[v] = lc1[v] * SCALE;
      cm = fmaxf(cm, fmaxf(x0[v], x1[v]));
    }
    cm = fmaxf(cm, swz_x16(cm));
    const float mnew  = fmaxf(m, cm);
    const float alpha = __builtin_amdgcn_exp2f((m - mnew) * LOG2E);
    float p0[8], p1[8], sc = 0.f;
    #pragma unroll
    for (int v = 0; v < 8; ++v) {
      p0[v] = __builtin_amdgcn_exp2f((x0[v] - mnew) * LOG2E);
      p1[v] = __builtin_amdgcn_exp2f((x1[v] - mnew) * LOG2E);
      sc += p0[v] + p1[v];
    }
    lsum = alpha * lsum + sc + swz_x16(sc);
    m = mnew;
    #pragma unroll
    for (int t = 0; t < 8; ++t) {
      #pragma unroll
      for (int v = 0; v < 8; ++v) acc[t][v] *= alpha;
    }

    // Build P^T B-fragment (32 s-rows x 16 lq-cols) from exp weights
    unsigned d0[4], d1[4];
    #pragma unroll
    for (int k = 0; k < 4; ++k) {
      d0[k] = pk_bf16(p0[2 * k], p0[2 * k + 1]);
      d1[k] = pk_bf16(p1[2 * k], p1[2 * k + 1]);
    }
    Frag pb;
    #pragma unroll
    for (int k = 0; k < 4; ++k) {
      unsigned sw0 = swz_x16u(d0[k]);
      unsigned sw1 = swz_x16u(d1[k]);
      pb.u[k]     = hl ? sw1   : d0[k];
      pb.u[4 + k] = hl ? d1[k] : sw0;
    }

    // ctx^T += V^T(16d x 32s) @ P^T(32s x 16lq), 8 d-chunks
    #pragma unroll
    for (int t = 0; t < 8; ++t) {
      const __bf16* vr = &VsBuf[buf][t * 16 + ln][16 * hl];
      acc[t] = WMMA_BF16(ld_frag2(vr, vr + 8), pb.v, acc[t]);
    }
    __syncthreads();                               // buf free for chunk ci+2
  }

  // normalize and store ctx[n][l][h*128 + d]; per t the 8 channels are
  // contiguous -> packed 16B store.
  const float rinv = 1.0f / lsum;
  __bf16* crow = Ctx + ((size_t)n * Lq + l0 + ln) * (size_t)(Hh * Dd) + h * Dd + 8 * hl;
  #pragma unroll
  for (int t = 0; t < 8; ++t) {
    uint4 w;
    w.x = pk_bf16(acc[t][0] * rinv, acc[t][1] * rinv);
    w.y = pk_bf16(acc[t][2] * rinv, acc[t][3] * rinv);
    w.z = pk_bf16(acc[t][4] * rinv, acc[t][5] * rinv);
    w.w = pk_bf16(acc[t][6] * rinv, acc[t][7] * rinv);
    *(uint4*)(crow + t * 16) = w;
  }
}

// ---------------------------------------------------------------------------
// Kernel 3: out = ctx[N,L,4096] @ Wc + bc  (fp32 output)
// One wave per 16x16 output tile, K=4096 contraction, 128 WMMAs.
// ---------------------------------------------------------------------------
__global__ void outproj_kernel(const __bf16* __restrict__ Ctx, const __bf16* __restrict__ Wct,
                               const float* __restrict__ bc, float* __restrict__ Out) {
  const int lane = threadIdx.x & 31, ln = lane & 15, hl = lane >> 4;
  const int wid = blockIdx.x * (blockDim.x >> 5) + (threadIdx.x >> 5);
  const int n = wid >> 10, rem = wid & 1023, l0 = (rem >> 3) * 16, c0 = (rem & 7) * 16;

  const __bf16* arow = Ctx + ((size_t)n * Lq + l0 + ln) * 4096;
  const __bf16* brow = Wct + (size_t)(c0 + ln) * 4096;

  v8f acc = {};
  for (int kc = 0; kc < 4096; kc += 128) {
    #pragma unroll
    for (int u = 0; u < 4; ++u) {
      const int k = kc + u * 32;
      v16bf a = ld_frag2(arow + k + 16 * hl, arow + k + 16 * hl + 8);
      v16bf bb = ld_frag2(brow + k + 16 * hl, brow + k + 16 * hl + 8);
      acc = WMMA_BF16(a, bb, acc);
    }
  }

  const float bias = bc[c0 + ln];
  float* o = Out + ((size_t)n * Lq + l0 + 8 * hl) * Dd + c0 + ln;
  #pragma unroll
  for (int v = 0; v < 8; ++v) o[(size_t)v * Dd] = acc[v] + bias;
}

// ---------------------------------------------------------------------------
extern "C" void kernel_launch(void* const* d_in, const int* in_sizes, int n_in,
                              void* d_out, int out_size, void* d_ws, size_t ws_size,
                              hipStream_t stream) {
  const float* q  = (const float*)d_in[0];
  const float* st = (const float*)d_in[1];
  const float* wk = (const float*)d_in[2];
  const float* bk = (const float*)d_in[3];
  const float* wv = (const float*)d_in[4];
  const float* bv = (const float*)d_in[5];
  const float* wc = (const float*)d_in[6];
  const float* bc = (const float*)d_in[7];
  float* out = (float*)d_out;

  char* ws = (char*)d_ws;
  const size_t MB = 1u << 20;
  __bf16* Qb  = (__bf16*)(ws + 0 * MB);    // [N][L][128]        1 MB
  __bf16* Sb  = (__bf16*)(ws + 1 * MB);    // [N][S][128]        1 MB
  __bf16* WkT = (__bf16*)(ws + 2 * MB);    // [H][k][d]          1 MB
  __bf16* WvT = (__bf16*)(ws + 3 * MB);    // [H][k][d]          1 MB
  __bf16* Wct = (__bf16*)(ws + 4 * MB);    // [128][4096]        1 MB
  __bf16* Kb  = (__bf16*)(ws + 5 * MB);    // [N][H][S][128]    32 MB
  __bf16* Vt  = (__bf16*)(ws + 37 * MB);   // [N][H][128][S]    32 MB
  __bf16* Ctx = (__bf16*)(ws + 69 * MB);   // [N][L][4096]      32 MB

  cvt_kernel<<<2048, 256, 0, stream>>>(q, st, wk, wv, wc, Qb, Sb, WkT, WvT, Wct);
  proj_kernel<<<2048, 256, 0, stream>>>(Sb, WkT, bk, Kb, 0);
  proj_kernel<<<2048, 256, 0, stream>>>(Sb, WvT, bv, Vt, 1);
  attn_kernel<<<1024, 256, 0, stream>>>(Qb, Kb, Vt, Ctx);
  outproj_kernel<<<256, 256, 0, stream>>>(Ctx, Wct, bc, out);
}
